// JointSupervisedGroundedCoreferencer_67242007986593
// MI455X (gfx1250) — compile-verified, matrix operand order checked by reading, not run on hardware
//
#include <hip/hip_runtime.h>
#include <hip/hip_bf16.h>
#include <math.h>

// ---------------- types ----------------
typedef __attribute__((ext_vector_type(16))) __bf16 v16bf;
typedef __attribute__((ext_vector_type(8)))  float  v8f;

#define B_    32
#define NT_   64
#define NS_   48
#define L_    36
#define D_    1024
#define H_    1024
#define NPAIR 1128          // 48*47/2
#define NEGV  (-9.0e9f)

__device__ __forceinline__ unsigned short f2bf(float x) {
  unsigned u = __float_as_uint(x);
  return (unsigned short)((u + 0x7FFFu + ((u >> 16) & 1u)) >> 16);  // RNE
}
__device__ __forceinline__ float bf2f(unsigned short h) {
  return __uint_as_float(((unsigned)h) << 16);
}
__device__ __forceinline__ v16bf ldfrag(const unsigned short* p) {
  return *(const v16bf*)(const void*)p;
}
__device__ __forceinline__ v8f wmma_bf16(v16bf a, v16bf b, v8f acc) {
  return __builtin_amdgcn_wmma_f32_16x16x32_bf16(false, a, false, b,
                                                 (short)0, acc, false, false);
}

// ---------------- prep: fp32 -> bf16 (weights transposed N x K), pair table ----------------
__global__ void k_prep(const float* __restrict__ text, const float* __restrict__ span,
                       const float* __restrict__ image,
                       const float* __restrict__ W1, const float* __restrict__ W2,
                       unsigned short* __restrict__ textBf, unsigned short* __restrict__ spanBf,
                       unsigned short* __restrict__ imageBf,
                       unsigned short* __restrict__ WT,   // W1aT,W1bT,W1cT,W2T back-to-back
                       int* __restrict__ pairIdx) {
  const long nText = (long)B_*NT_*D_;
  const long nSpan = (long)B_*NS_*D_;
  const long nImg  = (long)B_*L_*D_;
  const long nW    = 4L*H_*D_;
  const long total = nText + nSpan + nImg + nW + NPAIR;
  for (long idx = blockIdx.x * (long)blockDim.x + threadIdx.x; idx < total;
       idx += (long)gridDim.x * blockDim.x) {
    long t = idx;
    if (t < nText) { textBf[t]  = f2bf(text[t]);  continue; }
    t -= nText;
    if (t < nSpan) { spanBf[t]  = f2bf(span[t]);  continue; }
    t -= nSpan;
    if (t < nImg)  { imageBf[t] = f2bf(image[t]); continue; }
    t -= nImg;
    if (t < nW) {
      int w = (int)(t / ((long)H_*D_));
      int r = (int)(t % ((long)H_*D_));
      int n = r / D_, k = r % D_;
      float v = (w < 3) ? W1[(size_t)(k + w*D_)*H_ + n] : W2[(size_t)k*H_ + n];
      WT[t] = f2bf(v);
      continue;
    }
    t -= nW;
    { // triu_indices(48, k=1) order
      int p = (int)t, fi = 0, rem = p;
      while (rem >= (NS_ - 1 - fi)) { rem -= (NS_ - 1 - fi); ++fi; }
      pairIdx[2*p]     = fi;
      pairIdx[2*p + 1] = fi + 1 + rem;
    }
  }
}

// ---------------- generic batched bf16 WMMA GEMM, 2x2 register-blocked ----------------
// C[M,N] = A[M,K] * Bt[N,K]^T ; one wave computes a 32x32 output block (4 WMMA tiles)
__global__ void k_gemm(const unsigned short* __restrict__ A,
                       const unsigned short* __restrict__ Bt,
                       float* __restrict__ C,
                       int M, int N, int K,
                       long long sA, long long sB, long long sC) {
  int lane = threadIdx.x & 31, wave = threadIdx.x >> 5;
  int tilesN = (N + 31) >> 5, tilesM = (M + 31) >> 5;
  int waveId = blockIdx.x * (blockDim.x >> 5) + wave;
  if (waveId >= tilesM * tilesN) return;
  int mt = waveId / tilesN, nt = waveId % tilesN;
  int bz = blockIdx.y;
  int rsel = lane & 15, ksel = (lane >> 4) << 4;
  int m0 = mt << 5, n0 = nt << 5;
  const unsigned short* a0 = A  + (size_t)bz*sA + (size_t)(m0 + rsel)*K + ksel;
  const unsigned short* a1 = a0 + (size_t)16*K;
  const unsigned short* b0 = Bt + (size_t)bz*sB + (size_t)(n0 + rsel)*K + ksel;
  const unsigned short* b1 = b0 + (size_t)16*K;
  v8f c00 = {}, c01 = {}, c10 = {}, c11 = {};
  for (int kb = 0; kb < K; kb += 32) {
    v16bf fa0 = ldfrag(a0 + kb), fa1 = ldfrag(a1 + kb);
    v16bf fb0 = ldfrag(b0 + kb), fb1 = ldfrag(b1 + kb);
    c00 = wmma_bf16(fa0, fb0, c00);
    c01 = wmma_bf16(fa0, fb1, c01);
    c10 = wmma_bf16(fa1, fb0, c10);
    c11 = wmma_bf16(fa1, fb1, c11);
  }
  int col0  = n0 + rsel, col1 = col0 + 16;
  int mb0   = m0 + ((lane >> 4) << 3), mb1 = mb0 + 16;
  float* Cb = C + (size_t)bz*sC;
#pragma unroll
  for (int r = 0; r < 8; ++r) {
    int ma = mb0 + r, mbq = mb1 + r;
    if (ma < M) {
      if (col0 < N) Cb[(size_t)ma*N + col0] = c00[r];
      if (col1 < N) Cb[(size_t)ma*N + col1] = c01[r];
    }
    if (mbq < M) {
      if (col0 < N) Cb[(size_t)mbq*N + col0] = c10[r];
      if (col1 < N) Cb[(size_t)mbq*N + col1] = c11[r];
    }
  }
}

// ---------------- sentence score reduction: one block per (i,j) pair ----------------
__global__ void k_sent(const float* __restrict__ dotTI, const float* __restrict__ tmask,
                       const float* __restrict__ imask, float* __restrict__ S) {
  int i = blockIdx.x >> 5, j = blockIdx.x & 31;
  __shared__ float d[NT_][L_];
  __shared__ float fm[NT_], sm[L_];
  __shared__ float acc2[2];
  int tid = threadIdx.x;  // 128
  for (int e = tid; e < NT_*L_; e += 128) {
    int n = e / L_, l = e % L_;
    d[n][l] = dotTI[(size_t)(i*NT_ + n)*(B_*L_) + j*L_ + l];
  }
  if (tid < NT_) fm[tid] = tmask[i*NT_ + tid];
  if (tid < L_)  sm[tid] = imask[j*L_ + tid];
  if (tid < 2)   acc2[tid] = 0.f;
  __syncthreads();
  float cntN = 0.f, cntL = 0.f;
  for (int n = 0; n < NT_; ++n) cntN += fm[n];
  for (int l = 0; l < L_;  ++l) cntL += sm[l];
  if (tid < NT_) {            // row-wise softmax over l (term 1)
    int n = tid; float local = 0.f;
    if (fm[n] != 0.f) {
      float mx = -3.0e38f;
      for (int l = 0; l < L_; ++l) if (sm[l] != 0.f) mx = fmaxf(mx, d[n][l]);
      float s = 0.f, w = 0.f;
      for (int l = 0; l < L_; ++l) if (sm[l] != 0.f) {
        float e = expf(d[n][l] - mx); s += e; w += e * d[n][l];
      }
      local = (w / s) / cntN;
    }
    atomicAdd(&acc2[0], local);
  } else if (tid < NT_ + L_) { // column-wise softmax over n (term 2)
    int l = tid - NT_; float local = 0.f;
    if (sm[l] != 0.f) {
      float mx = -3.0e38f;
      for (int n = 0; n < NT_; ++n) if (fm[n] != 0.f) mx = fmaxf(mx, d[n][l]);
      float s = 0.f, w = 0.f;
      for (int n = 0; n < NT_; ++n) if (fm[n] != 0.f) {
        float e = expf(d[n][l] - mx); s += e; w += e * d[n][l];
      }
      local = (w / s) / cntL;
    }
    atomicAdd(&acc2[1], local);
  }
  __syncthreads();
  if (tid == 0) S[i*B_ + j] = acc2[0] + acc2[1];
}

// ---------------- contrastive loss from S (one wave) ----------------
__global__ void k_loss(const float* __restrict__ S, float* __restrict__ out) {
  __shared__ float sS[B_*B_];
  for (int t = threadIdx.x; t < B_*B_; t += 32) sS[t] = S[t];
  __syncthreads();
  int i = threadIdx.x;  // 0..31
  float mr = -3.0e38f, mc = -3.0e38f;
  for (int j = 0; j < B_; ++j) { mr = fmaxf(mr, sS[i*B_ + j]); mc = fmaxf(mc, sS[j*B_ + i]); }
  float sr = 0.f, sc = 0.f;
  for (int j = 0; j < B_; ++j) { sr += expf(sS[i*B_ + j] - mr); sc += expf(sS[j*B_ + i] - mc); }
  float diag = sS[i*B_ + i];
  float v = (diag - mr - logf(sr)) + (diag - mc - logf(sc));
  for (int o = 16; o > 0; o >>= 1) v += __shfl_down(v, o);
  if (i == 0) out[0] = -v / (float)B_;
}

// ---------------- grounding epilogue: mask ? dot : NEG ----------------
__global__ void k_ground(const float* __restrict__ dotSI, const float* __restrict__ smask,
                         const float* __restrict__ imask, float* __restrict__ out) {
  int idx = blockIdx.x * blockDim.x + threadIdx.x;
  if (idx >= B_*NS_*L_) return;
  int b = idx / (NS_*L_), r = idx % (NS_*L_);
  int n = r / L_, l = r % L_;
  float m = smask[b*NS_ + n] * imask[b*L_ + l];
  out[idx] = (m != 0.f) ? dotSI[idx] : NEGV;
}

// ---------------- fused pairwise MLP: 16 pairs per block, 8 waves ----------------
// A fragment (LDS activation tile) shared across 2 adjacent N tiles per K-step.
__global__ __launch_bounds__(256) void k_pairs(
    const float* __restrict__ span, const int* __restrict__ pairIdx,
    const float* __restrict__ PA, const float* __restrict__ PB,
    const unsigned short* __restrict__ W1cT, const unsigned short* __restrict__ W2T,
    const float* __restrict__ b1, const float* __restrict__ b2,
    const float* __restrict__ W3, const float* __restrict__ b3,
    float* __restrict__ out) {
  extern __shared__ unsigned short dynls[];          // 2 * 16*1024 bf16 = 64 KB
  unsigned short* bufA = dynls;                      // X3 tile, then h2 tile
  unsigned short* bufB = dynls + 16*1024;            // h1 tile
  __shared__ int   pf[16], ps[16];
  __shared__ float facc[16];
  int tid = threadIdx.x, bz = blockIdx.y, tile = blockIdx.x;
  if (tid < 16) {
    int p = tile*16 + tid;
    int q = (p < NPAIR) ? p : (NPAIR - 1);
    pf[tid] = pairIdx[2*q];
    ps[tid] = pairIdx[2*q + 1];
    facc[tid] = 0.f;
  }
  __syncthreads();
  // build X3 = first * second (bf16) for 16 pairs
  const float* sb = span + (size_t)bz * NS_ * D_;
  for (int e = tid; e < 16*D_; e += 256) {
    int pr = e >> 10, dd = e & (D_ - 1);
    bufA[e] = f2bf(sb[pf[pr]*D_ + dd] * sb[ps[pr]*D_ + dd]);
  }
  __syncthreads();
  int lane = tid & 31, wave = tid >> 5;
  int rsel = lane & 15, ksel = (lane >> 4) << 4;
  int mbase = (lane >> 4) << 3;
  // GEMM1: h1 = relu(X3 @ W1c + PA[fi] + PB[si] + b1); 2 N-tiles share the A frag
  for (int ntp = 0; ntp < 4; ++ntp) {
    int n0 = (wave*8 + ntp*2) << 4;
    const unsigned short* ap  = bufA + rsel*D_ + ksel;
    const unsigned short* bp0 = W1cT + (size_t)(n0 + rsel)*D_ + ksel;
    const unsigned short* bp1 = bp0 + (size_t)16*D_;
    v8f acc0 = {}, acc1 = {};
    for (int kb = 0; kb < D_; kb += 32) {
      v16bf av = ldfrag(ap + kb);
      acc0 = wmma_bf16(av, ldfrag(bp0 + kb), acc0);
      acc1 = wmma_bf16(av, ldfrag(bp1 + kb), acc1);
    }
    int col0 = n0 + rsel, col1 = col0 + 16;
#pragma unroll
    for (int r = 0; r < 8; ++r) {
      int m = mbase + r;
      size_t rowA = ((size_t)bz*NS_ + pf[m])*H_;
      size_t rowB = ((size_t)bz*NS_ + ps[m])*H_;
      float h0 = acc0[r] + PA[rowA + col0] + PB[rowB + col0] + b1[col0];
      float h1 = acc1[r] + PA[rowA + col1] + PB[rowB + col1] + b1[col1];
      bufB[m*H_ + col0] = f2bf(fmaxf(h0, 0.f));
      bufB[m*H_ + col1] = f2bf(fmaxf(h1, 0.f));
    }
  }
  __syncthreads();
  // GEMM2: h2 = relu(h1 @ W2 + b2)   (writes back into bufA)
  for (int ntp = 0; ntp < 4; ++ntp) {
    int n0 = (wave*8 + ntp*2) << 4;
    const unsigned short* ap  = bufB + rsel*H_ + ksel;
    const unsigned short* bp0 = W2T + (size_t)(n0 + rsel)*H_ + ksel;
    const unsigned short* bp1 = bp0 + (size_t)16*H_;
    v8f acc0 = {}, acc1 = {};
    for (int kb = 0; kb < H_; kb += 32) {
      v16bf av = ldfrag(ap + kb);
      acc0 = wmma_bf16(av, ldfrag(bp0 + kb), acc0);
      acc1 = wmma_bf16(av, ldfrag(bp1 + kb), acc1);
    }
    int col0 = n0 + rsel, col1 = col0 + 16;
#pragma unroll
    for (int r = 0; r < 8; ++r) {
      int m = mbase + r;
      bufA[m*H_ + col0] = f2bf(fmaxf(acc0[r] + b2[col0], 0.f));
      bufA[m*H_ + col1] = f2bf(fmaxf(acc1[r] + b2[col1], 0.f));
    }
  }
  __syncthreads();
  // final: out = h2 @ W3 + b3  (16 chunks of 64 per pair)
  {
    int pr = tid & 15, ch = tid >> 4;
    int d0 = ch * 64;
    float s = 0.f;
    for (int dd = 0; dd < 64; ++dd)
      s += bf2f(bufA[pr*H_ + d0 + dd]) * W3[d0 + dd];
    atomicAdd(&facc[pr], s);
  }
  __syncthreads();
  if (tid < 16) {
    int p = tile*16 + tid;
    if (p < NPAIR) out[(size_t)bz*NPAIR + p] = facc[tid] + b3[0];
  }
}

// ---------------- host launch ----------------
extern "C" void kernel_launch(void* const* d_in, const int* in_sizes, int n_in,
                              void* d_out, int out_size, void* d_ws, size_t ws_size,
                              hipStream_t stream) {
  const float* text  = (const float*)d_in[0];
  const float* span  = (const float*)d_in[1];
  const float* image = (const float*)d_in[2];
  const float* tmask = (const float*)d_in[3];
  const float* smask = (const float*)d_in[4];
  const float* imask = (const float*)d_in[5];
  const float* W1 = (const float*)d_in[6];
  const float* b1 = (const float*)d_in[7];
  const float* W2 = (const float*)d_in[8];
  const float* b2 = (const float*)d_in[9];
  const float* W3 = (const float*)d_in[10];
  const float* b3 = (const float*)d_in[11];
  float* out = (float*)d_out;

  // workspace carve-up (256B aligned)
  char* ws = (char*)d_ws;
  size_t off = 0;
  auto take = [&](size_t bytes) -> char* {
    char* p = ws + off;
    off = (off + bytes + 255) & ~(size_t)255;
    return p;
  };
  unsigned short* spanBf  = (unsigned short*)take((size_t)B_*NS_*D_*2);
  unsigned short* textBf  = (unsigned short*)take((size_t)B_*NT_*D_*2);
  unsigned short* imageBf = (unsigned short*)take((size_t)B_*L_*D_*2);
  unsigned short* WT      = (unsigned short*)take((size_t)4*H_*D_*2);  // W1aT,W1bT,W1cT,W2T
  unsigned short* W1aT = WT;
  unsigned short* W1bT = WT + (size_t)H_*D_;
  unsigned short* W1cT = WT + (size_t)2*H_*D_;
  unsigned short* W2T  = WT + (size_t)3*H_*D_;
  float* PA    = (float*)take((size_t)B_*NS_*H_*4);
  float* PB    = (float*)take((size_t)B_*NS_*H_*4);
  float* dotTI = (float*)take((size_t)(B_*NT_)*(B_*L_)*4);
  float* dotSI = (float*)take((size_t)B_*NS_*L_*4);
  float* Smat  = (float*)take((size_t)B_*B_*4);
  int*   pairIdx = (int*)take((size_t)NPAIR*2*4);

  // 1) convert embeddings/weights to bf16 (weights transposed N x K), build pair table
  k_prep<<<dim3(4096), dim3(256), 0, stream>>>(text, span, image, W1, W2,
                                               textBf, spanBf, imageBf, WT, pairIdx);
  // 2) PA = span @ W1a, PB = span @ W1b  (1536 x 1024 x 1024; 48x32 = 1536 32x32 blocks)
  k_gemm<<<dim3(192, 1), dim3(256), 0, stream>>>(
      spanBf, W1aT, PA, B_*NS_, H_, D_, 0, 0, 0);
  k_gemm<<<dim3(192, 1), dim3(256), 0, stream>>>(
      spanBf, W1bT, PB, B_*NS_, H_, D_, 0, 0, 0);
  // 3) dotTI = text_flat @ image_flat^T  (2048 x 1152 x 1024; 64x36 = 2304 blocks)
  k_gemm<<<dim3(288, 1), dim3(256), 0, stream>>>(
      textBf, imageBf, dotTI, B_*NT_, B_*L_, D_, 0, 0, 0);
  // 4) dotSI[b] = span[b] @ image[b]^T  (batched 48 x 36 x 1024; 2x2 = 4 blocks)
  k_gemm<<<dim3(1, B_), dim3(256), 0, stream>>>(
      spanBf, imageBf, dotSI, NS_, L_, D_,
      (long long)NS_*D_, (long long)L_*D_, (long long)NS_*L_);
  // 5) sentence scores + loss
  k_sent<<<dim3(B_*B_), dim3(128), 0, stream>>>(dotTI, tmask, imask, Smat);
  k_loss<<<dim3(1), dim3(32), 0, stream>>>(Smat, out);
  // 6) grounding scores (output slot 1 .. 1+B*Ns*L)
  k_ground<<<dim3((B_*NS_*L_ + 255)/256), dim3(256), 0, stream>>>(
      dotSI, smask, imask, out + 1);
  // 7) pairwise MLP -> text_scores (after loss + grounding)
  k_pairs<<<dim3((NPAIR + 15)/16, B_), dim3(256), 2*16*1024*sizeof(unsigned short), stream>>>(
      span, pairIdx, PA, PB, W1cT, W2T, b1, b2, W3, b3,
      out + 1 + (size_t)B_*NS_*L_);
}
